// EuclideanDistance_42666205118575
// MI455X (gfx1250) — matrix operand order repaired and checked by reading, not compiled, not run on hardware
//
#include <hip/hip_runtime.h>

// Squared Euclidean distance matrix: out = sq1 + sq2 - 2 * mat1 @ mat2^T
// MI455X (gfx1250): wave32, split-f16 WMMA (16x16x32, f32 accum), 320KB LDS/WGP.
//
// Cross term uses the Ootomo f16-split: a = a_hi + a_lo (f16 each),
// dot(a,b) ~= a_hi.b_hi + a_hi.b_lo + a_lo.b_hi accumulated in f32.
// 3 K=32 WMMAs replace 8 K=4 f32 WMMAs per K=32 -> ~2.7x fewer matrix ops,
// error ~2^-22 relative (negligible vs f32 reference).

typedef __attribute__((ext_vector_type(4)))  float    v4f;
typedef __attribute__((ext_vector_type(8)))  float    v8f;
typedef __attribute__((ext_vector_type(4)))  _Float16 v4h;
typedef __attribute__((ext_vector_type(8)))  _Float16 v8h;
typedef __attribute__((ext_vector_type(16))) _Float16 v16h;

#define N1 8192
#define N2 8192
#define DIM 128
#define TILE 128
#define SA 144  // A-tile LDS row stride (f16): dword stride 72 = 8 mod 64 -> conflict-free A frags
#define SB 136  // B-tile LDS row stride (f16): dword stride 68 = 4 mod 64 -> conflict-free B frags

// ---------------------------------------------------------------------------
// Kernel 1: per-row sum of squares in exact f32 (8 MB reads total, negligible)
// ---------------------------------------------------------------------------
__global__ __launch_bounds__(256) void rownorm_kernel(const float* __restrict__ m,
                                                      float* __restrict__ sq,
                                                      int nrows) {
  int row = blockIdx.x * blockDim.x + threadIdx.x;
  if (row >= nrows) return;
  const v4f* p = (const v4f*)(m + (size_t)row * DIM);
  float s = 0.0f;
#pragma unroll
  for (int i = 0; i < DIM / 4; ++i) {
    v4f v = p[i];
    s += v.x * v.x + v.y * v.y + v.z * v.z + v.w * v.w;
  }
  sq[row] = s;
}

// A-fragment per 16-bit WMMA lane map: lane holds K = g*8+{0..7} and 16+g*8+{0..7}
// -> two 16B runs stitched into one v16h.
__device__ __forceinline__ v16h frag_gap16(const _Float16* p) {
  v8h a = *(const v8h*)p;
  v8h b = *(const v8h*)(p + 16);
  return __builtin_shufflevector(a, b, 0, 1, 2, 3, 4, 5, 6, 7,
                                 8, 9, 10, 11, 12, 13, 14, 15);
}

// ---------------------------------------------------------------------------
// Kernel 2: 128x128 output tile per block, 8 wave32s (2x4), each wave a
// 64(M) x 32(N) patch = 4x2 tiles of v_wmma_f32_16x16x32_f16 (x3 split terms).
// ---------------------------------------------------------------------------
__global__ __launch_bounds__(256) void dist_tile_kernel(const float* __restrict__ A,
                                                        const float* __restrict__ B,
                                                        const float* __restrict__ sq1,
                                                        const float* __restrict__ sq2,
                                                        float* __restrict__ out) {
  __shared__ _Float16 sAhi[TILE * SA];
  __shared__ _Float16 sAlo[TILE * SA];
  __shared__ _Float16 sBhi[TILE * SB];
  __shared__ _Float16 sBlo[TILE * SB];

  const int i0 = blockIdx.y * TILE;
  const int j0 = blockIdx.x * TILE;
  const int tid = threadIdx.x;

  // Stage + split-convert: 32 consecutive threads stream one 512B row of each
  // operand (coalesced b128 loads), emit hi/lo f16 quads into LDS.
#pragma unroll
  for (int idx = tid; idx < TILE * (DIM / 4); idx += 256) {
    const int r = idx >> 5;
    const int c = (idx & 31) << 2;
    v4f va = *(const v4f*)(A + (size_t)(i0 + r) * DIM + c);
    v4f vb = *(const v4f*)(B + (size_t)(j0 + r) * DIM + c);
    v4h ah, al, bh, bl;
#pragma unroll
    for (int u = 0; u < 4; ++u) {
      _Float16 h = (_Float16)va[u];           // v_cvt_f16_f32 (RNE)
      ah[u] = h;
      al[u] = (_Float16)(va[u] - (float)h);   // residual, next 11-13 bits
      _Float16 g = (_Float16)vb[u];
      bh[u] = g;
      bl[u] = (_Float16)(vb[u] - (float)g);
    }
    *(v4h*)&sAhi[r * SA + c] = ah;
    *(v4h*)&sAlo[r * SA + c] = al;
    *(v4h*)&sBhi[r * SB + c] = bh;
    *(v4h*)&sBlo[r * SB + c] = bl;
  }
  __syncthreads();

  const int wave = tid >> 5;        // 0..7
  const int lane = tid & 31;
  const int ln = lane & 15;
  const int kgrp = lane >> 4;       // lane-half: K-group select per ISA layouts
  const int wm = (wave >> 2) * 64;  // wave M offset: 0 or 64
  const int wn = (wave & 3) * 32;   // wave N offset: 0,32,64,96

  v8f acc[4][2];
#pragma unroll
  for (int mi = 0; mi < 4; ++mi)
#pragma unroll
    for (int ni = 0; ni < 2; ++ni) acc[mi][ni] = {};

#pragma unroll
  for (int kb = 0; kb < DIM; kb += 32) {
    const int ab = kb + kgrp * 8;   // A lane K-run start (second run at +16)
    const int bb = kb + kgrp * 8;   // B lane K-run start (second run at +16)

    v16h Bh[2], Bl[2];
#pragma unroll
    for (int ni = 0; ni < 2; ++ni) {
      const int ro = (wn + ni * 16 + ln) * SB + bb;
      Bh[ni] = frag_gap16(&sBhi[ro]);
      Bl[ni] = frag_gap16(&sBlo[ro]);
    }
    v16h Ah[4], Al[4];
#pragma unroll
    for (int mi = 0; mi < 4; ++mi) {
      const int ro = (wm + mi * 16 + ln) * SA + ab;
      Ah[mi] = frag_gap16(&sAhi[ro]);
      Al[mi] = frag_gap16(&sAlo[ro]);
    }

#pragma unroll
    for (int mi = 0; mi < 4; ++mi)
#pragma unroll
      for (int ni = 0; ni < 2; ++ni) {
        acc[mi][ni] = __builtin_amdgcn_wmma_f32_16x16x32_f16(
            false, Ah[mi], false, Bh[ni], (short)0, acc[mi][ni], false, false);
        acc[mi][ni] = __builtin_amdgcn_wmma_f32_16x16x32_f16(
            false, Ah[mi], false, Bl[ni], (short)0, acc[mi][ni], false, false);
        acc[mi][ni] = __builtin_amdgcn_wmma_f32_16x16x32_f16(
            false, Al[mi], false, Bh[ni], (short)0, acc[mi][ni], false, false);
      }
  }

  // Epilogue: C/D layout -> VGPR v holds M = v + 8*kgrp, N = ln.
#pragma unroll
  for (int mi = 0; mi < 4; ++mi) {
#pragma unroll
    for (int ni = 0; ni < 2; ++ni) {
      const int n = j0 + wn + ni * 16 + ln;
      const float s2 = sq2[n];
#pragma unroll
      for (int v = 0; v < 8; ++v) {
        const int m = i0 + wm + mi * 16 + v + kgrp * 8;
        out[(size_t)m * N2 + n] = sq1[m] + s2 - 2.0f * acc[mi][ni][v];
      }
    }
  }
}

// ---------------------------------------------------------------------------
extern "C" void kernel_launch(void* const* d_in, const int* in_sizes, int n_in,
                              void* d_out, int out_size, void* d_ws, size_t ws_size,
                              hipStream_t stream) {
  (void)in_sizes; (void)n_in; (void)out_size; (void)ws_size;
  const float* mat1 = (const float*)d_in[0];
  const float* mat2 = (const float*)d_in[1];
  float* out = (float*)d_out;
  float* sq1 = (float*)d_ws;  // N1 floats
  float* sq2 = sq1 + N1;      // N2 floats

  rownorm_kernel<<<N1 / 256, 256, 0, stream>>>(mat1, sq1, N1);
  rownorm_kernel<<<N2 / 256, 256, 0, stream>>>(mat2, sq2, N2);

  dim3 grid(N2 / TILE, N1 / TILE);  // 64 x 64 blocks
  dist_tile_kernel<<<grid, 256, 0, stream>>>(mat1, mat2, sq1, sq2, out);
}
